// LSTMCell_59425167507716
// MI455X (gfx1250) — matrix operand order repaired and checked by reading, not compiled, not run on hardware
//
#include <hip/hip_runtime.h>

typedef __attribute__((ext_vector_type(16))) _Float16 v16h;
typedef __attribute__((ext_vector_type(8)))  _Float16 v8h;
typedef __attribute__((ext_vector_type(8)))  float    v8f;

#define BATCH  131072
#define INDIM  100
#define HD     256
#define KX     128          // x K-extent padded to 128
#define KDIM   384          // 128 (x, padded) + 256 (h)
#define LDK    392          // LDS K stride (padding to spread banks)
#define BM     64           // rows per block
#define BN     32           // n-columns per block (x4 gates = 128 gate cols)

// ---------------------------------------------------------------------------
// Prep: Wt[n][k] (f16, n = gate column 0..1023, k = 0..383) and bsum = bx+bh.
// Wt is "transposed" so B-matrix fragments read two contiguous 8-half K runs per lane.
// ---------------------------------------------------------------------------
__global__ __launch_bounds__(256) void lstm_prep(
    const float* __restrict__ Wx, const float* __restrict__ Wh,
    const float* __restrict__ bx, const float* __restrict__ bh,
    _Float16* __restrict__ Wt, float* __restrict__ bsum)
{
    int idx = blockIdx.x * 256 + threadIdx.x;     // 1024*384 total
    int n = idx / KDIM;
    int k = idx % KDIM;
    float v;
    if (k < INDIM)      v = Wx[(size_t)k * 1024 + n];
    else if (k < KX)    v = 0.0f;
    else                v = Wh[(size_t)(k - KX) * 1024 + n];
    Wt[(size_t)n * KDIM + k] = (_Float16)v;
    if (k == 0) bsum[n] = bx[n] + bh[n];
}

// ---------------------------------------------------------------------------
// Fused GEMM (f16 WMMA, f32 accum) + LSTM epilogue.
// Block: 256 thr = 8 waves = 4 rowTiles x 2 nTiles; each wave holds all 4
// gate accumulators for its 16x16 tile so the epilogue stays in registers.
// ---------------------------------------------------------------------------
__global__ __launch_bounds__(256) void lstm_gemm(
    const float* __restrict__ x, const float* __restrict__ C,
    const float* __restrict__ h,
    const _Float16* __restrict__ Wt, const float* __restrict__ bsum,
    float* __restrict__ out)
{
    __shared__ __align__(16) _Float16 Ash[BM][LDK];        // 64  x 392 f16
    __shared__ __align__(16) _Float16 Bsh[4 * BN][LDK];    // 128 x 392 f16

    const int tid  = threadIdx.x;
    const int row0 = blockIdx.y * BM;
    const int n0   = blockIdx.x * BN;

    // --- stage A: x part (K 0..127, zero-padded past 99), f32 -> f16
    for (int i = tid; i < BM * KX; i += 256) {
        int r = i >> 7;
        int k = i & (KX - 1);
        float v = (k < INDIM) ? x[(size_t)(row0 + r) * INDIM + k] : 0.0f;
        Ash[r][k] = (_Float16)v;
    }
    // --- stage A: h part (K 128..383)
    for (int i = tid; i < BM * HD; i += 256) {
        int r = i >> 8;
        int k = i & (HD - 1);
        Ash[r][KX + k] = (_Float16)h[(size_t)(row0 + r) * HD + k];
    }
    // --- stage B: 128 gate columns x 384 K halfs, 16B vector copies from L2
    for (int i = tid; i < (4 * BN) * (KDIM / 8); i += 256) {
        int c  = i / (KDIM / 8);
        int k8 = i % (KDIM / 8);
        int gcol = (c >> 5) * HD + n0 + (c & 31);          // gate*256 + n
        *(v8h*)&Bsh[c][k8 * 8] =
            *(const v8h*)(Wt + (size_t)gcol * KDIM + k8 * 8);
    }
    __syncthreads();

    const int wave    = tid >> 5;
    const int lane    = tid & 31;
    const int rowTile = wave & 3;      // 0..3
    const int nTile   = wave >> 2;     // 0..1
    const int lhalf   = lane >> 4;     // 0/1
    const int l16     = lane & 15;

    v8f acc[4] = {};                   // i, f, g, o accumulators (f32)

    const _Float16* arow = &Ash[rowTile * 16 + l16][0];
    #pragma unroll
    for (int ks = 0; ks < KDIM / 32; ++ks) {
        const int kb = ks * 32;
        // A fragment (16x32 f16): VGPR0-3 = K kb+8h..+7, VGPR4-7 = K kb+16+8h..
        v8h alo = *(const v8h*)(arow + kb + lhalf * 8);
        v8h ahi = *(const v8h*)(arow + kb + 16 + lhalf * 8);
        v16h a = __builtin_shufflevector(alo, ahi,
                 0,1,2,3,4,5,6,7,8,9,10,11,12,13,14,15);
        #pragma unroll
        for (int g = 0; g < 4; ++g) {
            const _Float16* bcol = &Bsh[g * 32 + nTile * 16 + l16][0];
            // B fragment (32x16 f16): VGPR0-3 = K kb+8h..+7, VGPR4-7 = K kb+16+8h..
            v8h blo = *(const v8h*)(bcol + kb + lhalf * 8);
            v8h bhi = *(const v8h*)(bcol + kb + 16 + lhalf * 8);
            v16h b = __builtin_shufflevector(blo, bhi,
                     0,1,2,3,4,5,6,7,8,9,10,11,12,13,14,15);
            acc[g] = __builtin_amdgcn_wmma_f32_16x16x32_f16(
                         false, a, false, b, (short)0, acc[g], false, false);
        }
    }

    // --- epilogue: bias + activations + cell update, all in registers
    const int ncol = n0 + nTile * 16 + l16;                // 0..255
    const float bi = bsum[ncol];
    const float bf = bsum[HD + ncol];
    const float bg = bsum[2 * HD + ncol];
    const float bo = bsum[3 * HD + ncol];
    const int rowbase = row0 + rowTile * 16 + lhalf * 8;   // C/D layout: M=8h+r

    #pragma unroll
    for (int r = 0; r < 8; ++r) {
        const size_t row = (size_t)(rowbase + r);
        float iv = 1.0f / (1.0f + __expf(-(acc[0][r] + bi)));
        float fv = 1.0f / (1.0f + __expf(-(acc[1][r] + bf)));
        float gv = tanhf(acc[2][r] + bg);
        float ov = 1.0f / (1.0f + __expf(-(acc[3][r] + bo)));
        float cin = C[row * HD + ncol];
        float co  = fv * cin + iv * gv;
        float ho  = ov * tanhf(co);
        out[row * HD + ncol]                        = co;   // c_out
        out[(size_t)BATCH * HD + row * HD + ncol]   = ho;   // h_out
    }
}

// ---------------------------------------------------------------------------
extern "C" void kernel_launch(void* const* d_in, const int* in_sizes, int n_in,
                              void* d_out, int out_size, void* d_ws, size_t ws_size,
                              hipStream_t stream) {
    const float* x  = (const float*)d_in[0];
    const float* C  = (const float*)d_in[1];
    const float* h  = (const float*)d_in[2];
    const float* Wx = (const float*)d_in[3];
    const float* Wh = (const float*)d_in[4];
    const float* bx = (const float*)d_in[5];
    const float* bh = (const float*)d_in[6];
    float* out = (float*)d_out;

    _Float16* Wt = (_Float16*)d_ws;                                 // 1024*384 f16
    float* bsum  = (float*)((char*)d_ws + (size_t)1024 * KDIM * 2); // 1024 f32

    lstm_prep<<<(1024 * KDIM) / 256, 256, 0, stream>>>(Wx, Wh, bx, bh, Wt, bsum);

    dim3 grid(HD / BN, BATCH / BM);   // (8 col-blocks, 2048 row-blocks)
    lstm_gemm<<<grid, 256, 0, stream>>>(x, C, h, Wt, bsum, out);
}
